// MMSingleStreamBlock_55173149884873
// MI455X (gfx1250) — compile-verified, matrix operand order checked by reading, not run on hardware
//
#include <hip/hip_runtime.h>
#include <stdint.h>

// ---------------- problem constants ----------------
#define S_TOT 2304        // LI + LT
#define LI    2048
#define LT    256
#define HID   3072
#define HEADS 24
#define HD    128
#define MLP   12288
#define NQKV  (3*HID)     // 9216
#define N1    (3*HID+MLP) // 21504
#define EPSF  1e-6f

typedef unsigned short u16;
typedef __attribute__((ext_vector_type(16))) __bf16 v16bf;
typedef __attribute__((ext_vector_type(8)))  float  v8f;

// ---------------- bf16 helpers ----------------
__device__ __forceinline__ u16 f2bf(float f) {
  union { float f; uint32_t u; } v; v.f = f;
  uint32_t u = v.u;
  u += 0x7FFFu + ((u >> 16) & 1u);   // round-to-nearest-even
  return (u16)(u >> 16);
}
__device__ __forceinline__ float bf2f(u16 h) {
  union { uint32_t u; float f; } v; v.u = ((uint32_t)h) << 16;
  return v.f;
}

// ---------------- wave reductions (wave32) ----------------
__device__ __forceinline__ float rsum32(float v) {
#pragma unroll
  for (int m = 1; m < 32; m <<= 1) v += __shfl_xor(v, m, 32);
  return v;
}
__device__ __forceinline__ float rsum16(float v) {   // within 16-lane half
#pragma unroll
  for (int m = 1; m < 16; m <<= 1) v += __shfl_xor(v, m, 32);
  return v;
}
__device__ __forceinline__ float rmax16(float v) {
#pragma unroll
  for (int m = 1; m < 16; m <<= 1) v = fmaxf(v, __shfl_xor(v, m, 32));
  return v;
}

// ---------------- WMMA bf16 fragment loader ----------------
// ISA 7.12.2 16-bit A 16x32: lanes 0-15 row=lane hold K=0..7,16..23;
// lanes 16-31 same rows hold K=8..15,24..31. B (W row-major [N,K]) mirrors
// with N=lane&15. ld in elements; all offsets 16B aligned.
__device__ __forceinline__ v16bf load_frag(const u16* base, int ld) {
  const int lane = threadIdx.x & 31;
  const int r    = lane & 15;
  const int hf   = lane >> 4;
  const u16* p = base + (size_t)r * ld + hf * 8;
  v16bf f;
  *((uint4*)&f)       = *((const uint4*)p);        // K = 8*hf .. +7
  *(((uint4*)&f) + 1) = *((const uint4*)(p + 16)); // K = 16+8*hf .. +7
  return f;
}
#define WMMA_BF16(A,B,C) __builtin_amdgcn_wmma_f32_16x16x32_bf16(false,(A),false,(B),(short)0,(C),false,false)

// ---------------- CDNA5 async global->LDS copy (ASYNCcnt path) ----------------
// VDST = per-lane LDS byte address (low 32 bits of generic shared pointer,
// ISA 10.2: LDS aperture keeps the wave-relative offset in addr[31:0]).
__device__ __forceinline__ void async_b128(const u16* lds_dst, const u16* gsrc) {
  unsigned la = (unsigned)(size_t)lds_dst;
  asm volatile("global_load_async_to_lds_b128 %0, %1, off"
               :: "v"(la), "v"(gsrc) : "memory");
}
__device__ __forceinline__ void wait_async_le8() {
  asm volatile("s_wait_asynccnt 8" ::: "memory");
}
__device__ __forceinline__ void wait_async_0() {
  asm volatile("s_wait_asynccnt 0" ::: "memory");
}

// ---------------- K0: fp32 -> bf16 convert ----------------
__global__ void cvt_bf16_kernel(const float* __restrict__ s, u16* __restrict__ d, size_t n) {
  size_t i = (size_t)blockIdx.x * blockDim.x + threadIdx.x;
  if (i < n) d[i] = f2bf(s[i]);
}

// ---------------- K1a: silu(vec) ----------------
__global__ void silu_kernel(const float* __restrict__ vec, float* __restrict__ sv) {
  int i = blockIdx.x * blockDim.x + threadIdx.x;
  if (i < HID) { float x = vec[i]; sv[i] = x / (1.0f + __expf(-x)); }
}

// ---------------- K1b: modulation GEMV: mod = silu(vec) @ mod_w^T + mod_b ----------------
__global__ __launch_bounds__(256) void mod_gemv_kernel(const float* __restrict__ sv,
                                                       const float* __restrict__ mw,
                                                       const float* __restrict__ mb,
                                                       float* __restrict__ mod) {
  int gw   = (blockIdx.x * 256 + threadIdx.x) >> 5;  // one wave per output
  int lane = threadIdx.x & 31;
  if (gw >= NQKV) return;
  const float* wrow = mw + (size_t)gw * HID;
  float acc = 0.f;
  for (int k = lane; k < HID; k += 32) acc += sv[k] * wrow[k];
  acc = rsum32(acc);
  if (lane == 0) mod[gw] = acc + mb[gw];
}

// ---------------- K2: LayerNorm + modulate -> x_mod (bf16) ----------------
__global__ __launch_bounds__(256) void ln_mod_kernel(const float* __restrict__ img,
                                                     const float* __restrict__ txt,
                                                     const float* __restrict__ mod,
                                                     u16* __restrict__ xmod) {
  const int row = blockIdx.x;
  const float* x = (row < LI) ? img + (size_t)row * HID : txt + (size_t)(row - LI) * HID;
  __shared__ float red[16];
  float s = 0.f, s2 = 0.f;
  for (int c = threadIdx.x; c < HID; c += 256) { float v = x[c]; s += v; s2 += v * v; }
  s = rsum32(s); s2 = rsum32(s2);
  int wave = threadIdx.x >> 5, lane = threadIdx.x & 31;
  if (lane == 0) { red[wave] = s; red[8 + wave] = s2; }
  __syncthreads();
  if (threadIdx.x == 0) {
    float a = 0.f, b = 0.f;
#pragma unroll
    for (int w = 0; w < 8; ++w) { a += red[w]; b += red[8 + w]; }
    red[0] = a; red[1] = b;
  }
  __syncthreads();
  float mu  = red[0] * (1.0f / HID);
  float var = red[1] * (1.0f / HID) - mu * mu;
  float inv = rsqrtf(var + EPSF);
  for (int c = threadIdx.x; c < HID; c += 256) {
    float v = (x[c] - mu) * inv;
    v = v * (1.0f + mod[HID + c]) + mod[c];      // scale, shift
    xmod[(size_t)row * HID + c] = f2bf(v);
  }
}

// ---------------- K3/K8: bf16 WMMA GEMM  C[M,N] = A[M,K] @ W[N,K]^T + bias ----------------
// Double-buffered LDS, async global->LDS loads overlap next tile with WMMAs.
// mode 0: store f32 C.  mode 1: out = residual + gate*(C+bias) -> d_out.
#define BM 128
#define BN 128
#define BK 64
#define LDP (BK + 8)   // padded LDS row stride (144B: 16B-aligned, bank-skewed)

__global__ __launch_bounds__(256) void gemm_bf16_kernel(
    const u16* __restrict__ A, const u16* __restrict__ Bw,
    int M, int N, int K,
    const float* __restrict__ bias, float* __restrict__ C, int mode,
    const float* __restrict__ img, const float* __restrict__ txt,
    const float* __restrict__ gate, float* __restrict__ outp) {
  __shared__ u16 As[2][BM * LDP];
  __shared__ u16 Bs[2][BN * LDP];
  const int tid  = threadIdx.x;
  const int lane = tid & 31;
  const int wave = tid >> 5;
  const int wm = wave & 3;          // 4 waves over M -> 32-row strip each
  const int wn = wave >> 2;         // 2 waves over N -> 64-col strip each
  const int tileM = (int)blockIdx.x * BM;
  const int tileN = (int)blockIdx.y * BN;

  // this thread's slice of the tile copy: 4 16B vectors for A, 4 for B
  int vrow[4], vcol[4];
#pragma unroll
  for (int t = 0; t < 4; ++t) {
    int vec = tid + t * 256;        // 0..1023
    vrow[t] = vec >> 3;
    vcol[t] = (vec & 7) * 8;
  }
  const int nk = K / BK;

  // prologue: issue group 0 into buffer 0
#pragma unroll
  for (int t = 0; t < 4; ++t) {
    async_b128(&As[0][vrow[t] * LDP + vcol[t]],
               &A [(size_t)(tileM + vrow[t]) * K + vcol[t]]);
    async_b128(&Bs[0][vrow[t] * LDP + vcol[t]],
               &Bw[(size_t)(tileN + vrow[t]) * K + vcol[t]]);
  }

  v8f acc[2][4] = {};

  for (int i = 0; i < nk; ++i) {
    if (i + 1 < nk) {               // issue next group into the other buffer
      const int k0 = (i + 1) * BK, b = (i + 1) & 1;
#pragma unroll
      for (int t = 0; t < 4; ++t) {
        async_b128(&As[b][vrow[t] * LDP + vcol[t]],
                   &A [(size_t)(tileM + vrow[t]) * K + k0 + vcol[t]]);
        async_b128(&Bs[b][vrow[t] * LDP + vcol[t]],
                   &Bw[(size_t)(tileN + vrow[t]) * K + k0 + vcol[t]]);
      }
      wait_async_le8();             // group i complete (in-order) ...
    } else {
      wait_async_0();
    }
    __syncthreads();                // ... for every wave in the block

    const u16* Asb = As[i & 1];
    const u16* Bsb = Bs[i & 1];
#pragma unroll
    for (int kk = 0; kk < BK; kk += 32) {
      v16bf af0 = load_frag(&Asb[(wm * 32 +  0) * LDP + kk], LDP);
      v16bf af1 = load_frag(&Asb[(wm * 32 + 16) * LDP + kk], LDP);
#pragma unroll
      for (int j = 0; j < 4; ++j) {
        v16bf bf = load_frag(&Bsb[(wn * 64 + j * 16) * LDP + kk], LDP);
        acc[0][j] = WMMA_BF16(af0, bf, acc[0][j]);
        acc[1][j] = WMMA_BF16(af1, bf, acc[1][j]);
      }
    }
    __syncthreads();                // buffer i&1 free before refill at i+2
  }

  // epilogue (C layout: VGPR r -> row r (lanes 0-15) / r+8 (lanes 16-31))
  const int hf = lane >> 4, c0 = lane & 15;
#pragma unroll
  for (int i = 0; i < 2; ++i)
#pragma unroll
    for (int j = 0; j < 4; ++j)
#pragma unroll
      for (int r = 0; r < 8; ++r) {
        int row = tileM + wm * 32 + i * 16 + hf * 8 + r;
        int col = tileN + wn * 64 + j * 16 + c0;
        float v = acc[i][j][r] + bias[col];
        if (mode == 0) {
          C[(size_t)row * N + col] = v;
        } else {
          float res = (row < LI) ? img[(size_t)row * HID + col]
                                 : txt[(size_t)(row - LI) * HID + col];
          outp[(size_t)row * HID + col] = res + gate[col] * v;
        }
      }
}

// ---------------- K4: QKV postprocess: RMSNorm + RoPE, head-major bf16 ----------------
__global__ __launch_bounds__(256) void qkv_post_kernel(
    const float* __restrict__ y1, const float* __restrict__ qw, const float* __restrict__ kw,
    const float* __restrict__ fcos, const float* __restrict__ fsin,
    u16* __restrict__ qb, u16* __restrict__ kb, u16* __restrict__ vT) {
  int gw   = (int)((blockIdx.x * 256 + threadIdx.x) >> 5);  // one wave per (s,h)
  int lane = threadIdx.x & 31;
  int s = gw / HEADS, h = gw % HEADS;
  if (s >= S_TOT) return;
  const float* base = y1 + (size_t)s * N1;
  const int d0 = lane * 4;
  float q[4], k[4], v[4];
#pragma unroll
  for (int t = 0; t < 4; ++t) {
    q[t] = base[          h * HD + d0 + t];
    k[t] = base[HID     + h * HD + d0 + t];
    v[t] = base[2 * HID + h * HD + d0 + t];
  }
  float sq = 0.f, sk = 0.f;
#pragma unroll
  for (int t = 0; t < 4; ++t) { sq += q[t] * q[t]; sk += k[t] * k[t]; }
  sq = rsum32(sq); sk = rsum32(sk);
  float rq = rsqrtf(sq * (1.0f / HD) + EPSF);
  float rk = rsqrtf(sk * (1.0f / HD) + EPSF);
#pragma unroll
  for (int t = 0; t < 4; ++t) { q[t] *= rq * qw[d0 + t]; k[t] *= rk * kw[d0 + t]; }
  if (s < LI) {                               // RoPE only on img tokens
#pragma unroll
    for (int p = 0; p < 2; ++p) {
      int d = d0 + 2 * p;
      float c = fcos[(size_t)s * HD + d], sn = fsin[(size_t)s * HD + d];
      float q0 = q[2 * p], q1 = q[2 * p + 1];
      q[2 * p] = q0 * c - q1 * sn;  q[2 * p + 1] = q1 * c + q0 * sn;
      float k0 = k[2 * p], k1 = k[2 * p + 1];
      k[2 * p] = k0 * c - k1 * sn;  k[2 * p + 1] = k1 * c + k0 * sn;
    }
  }
#pragma unroll
  for (int t = 0; t < 4; ++t) {
    qb[(size_t)h * S_TOT * HD + (size_t)s * HD + d0 + t] = f2bf(q[t]);
    kb[(size_t)h * S_TOT * HD + (size_t)s * HD + d0 + t] = f2bf(k[t]);
    vT[(size_t)h * HD * S_TOT + (size_t)(d0 + t) * S_TOT + s] = f2bf(v[t]); // d-major
  }
}

// ---------------- K5: gelu-tanh -> cat[:, 3072:] ----------------
__global__ void gelu_kernel(const float* __restrict__ y1, u16* __restrict__ cat) {
  size_t i = (size_t)blockIdx.x * blockDim.x + threadIdx.x;
  if (i >= (size_t)S_TOT * MLP) return;
  size_t s = i / MLP; int c = (int)(i % MLP);
  float x = y1[s * N1 + NQKV + c];
  float g = 0.5f * x * (1.0f + tanhf(0.7978845608028654f * (x + 0.044715f * x * x * x)));
  cat[s * N1 + HID + c] = f2bf(g);
}

// ---------------- K6: flash attention (bf16 WMMA, online softmax) ----------------
// grid (HEADS, S/128); 8 waves, each owns 16 query rows; kv step 32.
#define PLD 40   // padded P-tile row stride (80B, 16B aligned)
__global__ __launch_bounds__(256) void attn_kernel(
    const u16* __restrict__ qb, const u16* __restrict__ kb,
    const u16* __restrict__ vT, u16* __restrict__ cat) {
  __shared__ u16 pbuf[8][16 * PLD];
  const int lane = threadIdx.x & 31;
  const int wave = threadIdx.x >> 5;
  const int hf = lane >> 4, c0 = lane & 15;
  const int h  = blockIdx.x;
  const int q0 = (int)blockIdx.y * 128 + wave * 16;
  const float scl = 0.08838834764831845f;   // 1/sqrt(128)

  const u16* Q  = qb + (size_t)h * S_TOT * HD;
  const u16* Kp = kb + (size_t)h * S_TOT * HD;
  const u16* Vp = vT + (size_t)h * HD * S_TOT;
  u16* pw = &pbuf[wave][0];

  v16bf qf[4];
#pragma unroll
  for (int d = 0; d < 4; ++d) qf[d] = load_frag(Q + (size_t)q0 * HD + d * 32, HD);

  v8f o[8] = {};
  float m[8], l[8];
#pragma unroll
  for (int r = 0; r < 8; ++r) { m[r] = -1e30f; l[r] = 0.f; }

  for (int kv = 0; kv < S_TOT; kv += 32) {
    v8f s0 = {}, s1 = {};
#pragma unroll
    for (int d = 0; d < 4; ++d) {
      v16bf kf0 = load_frag(Kp + (size_t)kv * HD + d * 32, HD);
      v16bf kf1 = load_frag(Kp + (size_t)(kv + 16) * HD + d * 32, HD);
      s0 = WMMA_BF16(qf[d], kf0, s0);
      s1 = WMMA_BF16(qf[d], kf1, s1);
    }
    // online softmax per row; a row's 16 cols live in one 16-lane half
#pragma unroll
    for (int r = 0; r < 8; ++r) {
      float a = s0[r] * scl, b = s1[r] * scl;
      float mx = rmax16(fmaxf(a, b));
      float mnew = fmaxf(m[r], mx);
      float alpha = __expf(m[r] - mnew);
      float p0 = __expf(a - mnew), p1 = __expf(b - mnew);
      float rs = rsum16(p0 + p1);
      l[r] = l[r] * alpha + rs;
      m[r] = mnew;
#pragma unroll
      for (int j = 0; j < 8; ++j) o[j][r] *= alpha;
      int row = r + 8 * hf;
      pw[row * PLD + c0]      = f2bf(p0);
      pw[row * PLD + 16 + c0] = f2bf(p1);
    }
    asm volatile("s_wait_dscnt 0" ::: "memory");   // LDS RAW within wave (CDNA5 split counter)
    v16bf pf = load_frag(pw, PLD);                 // P as A-fragment (16x32)
#pragma unroll
    for (int j = 0; j < 8; ++j) {
      v16bf vf = load_frag(Vp + (size_t)(j * 16) * S_TOT + kv, S_TOT);
      o[j] = WMMA_BF16(pf, vf, o[j]);
    }
  }
  // write attn output into cat[:, h*128 + d]
#pragma unroll
  for (int j = 0; j < 8; ++j)
#pragma unroll
    for (int r = 0; r < 8; ++r) {
      int row = q0 + r + 8 * hf;
      int col = h * HD + j * 16 + c0;
      cat[(size_t)row * N1 + col] = f2bf(o[j][r] / l[r]);
    }
}

// ---------------- host-side launcher ----------------
extern "C" void kernel_launch(void* const* d_in, const int* in_sizes, int n_in,
                              void* d_out, int out_size, void* d_ws, size_t ws_size,
                              hipStream_t stream) {
  (void)in_sizes; (void)n_in; (void)out_size; (void)ws_size;
  const float* img  = (const float*)d_in[0];
  const float* txt  = (const float*)d_in[1];
  const float* vec  = (const float*)d_in[2];
  const float* fcos = (const float*)d_in[3];
  const float* fsin = (const float*)d_in[4];
  const float* w1   = (const float*)d_in[5];
  const float* b1   = (const float*)d_in[6];
  const float* w2   = (const float*)d_in[7];
  const float* b2   = (const float*)d_in[8];
  const float* qw   = (const float*)d_in[9];
  const float* kw   = (const float*)d_in[10];
  const float* mw   = (const float*)d_in[11];
  const float* mb   = (const float*)d_in[12];
  float* outp = (float*)d_out;

  size_t off = 0;
  char* ws = (char*)d_ws;
  auto carve = [&](size_t bytes) -> void* {
    void* p = ws + off; off += (bytes + 255) & ~(size_t)255; return p;
  };
  u16*   w1b  = (u16*)  carve((size_t)N1 * HID * 2);
  u16*   w2b  = (u16*)  carve((size_t)HID * N1 * 2);
  u16*   xmod = (u16*)  carve((size_t)S_TOT * HID * 2);
  float* y1   = (float*)carve((size_t)S_TOT * N1 * 4);
  u16*   qbb  = (u16*)  carve((size_t)HEADS * S_TOT * HD * 2);
  u16*   kbb  = (u16*)  carve((size_t)HEADS * S_TOT * HD * 2);
  u16*   vTb  = (u16*)  carve((size_t)HEADS * HD * S_TOT * 2);
  u16*   cat  = (u16*)  carve((size_t)S_TOT * N1 * 2);
  float* sv   = (float*)carve((size_t)HID * 4);
  float* modv = (float*)carve((size_t)NQKV * 4);

  const size_t nw = (size_t)N1 * HID;
  cvt_bf16_kernel<<<(unsigned)((nw + 255) / 256), 256, 0, stream>>>(w1, w1b, nw);
  cvt_bf16_kernel<<<(unsigned)((nw + 255) / 256), 256, 0, stream>>>(w2, w2b, nw);
  silu_kernel<<<(HID + 255) / 256, 256, 0, stream>>>(vec, sv);
  mod_gemv_kernel<<<NQKV / 8, 256, 0, stream>>>(sv, mw, mb, modv);
  ln_mod_kernel<<<S_TOT, 256, 0, stream>>>(img, txt, modv, xmod);
  gemm_bf16_kernel<<<dim3(S_TOT / BM, N1 / BN), 256, 0, stream>>>(
      xmod, w1b, S_TOT, N1, HID, b1, y1, 0, nullptr, nullptr, nullptr, nullptr);
  qkv_post_kernel<<<(S_TOT * HEADS) / 8, 256, 0, stream>>>(
      y1, qw, kw, fcos, fsin, qbb, kbb, vTb);
  const size_t ng = (size_t)S_TOT * MLP;
  gelu_kernel<<<(unsigned)((ng + 255) / 256), 256, 0, stream>>>(y1, cat);
  attn_kernel<<<dim3(HEADS, S_TOT / 128), 256, 0, stream>>>(qbb, kbb, vTb, cat);
  gemm_bf16_kernel<<<dim3(S_TOT / BM, HID / BN), 256, 0, stream>>>(
      cat, w2b, S_TOT, HID, N1, b2, nullptr, 1, img, txt, modv + 2 * HID, outp);
}
